// BiLSTM_CRF_NER_67405216744272
// MI455X (gfx1250) — matrix operand order repaired and checked by reading, not compiled, not run on hardware
//
#include <hip/hip_runtime.h>
#include <stdint.h>

typedef __attribute__((ext_vector_type(16))) __bf16 v16bf;
typedef __attribute__((ext_vector_type(8)))  float  v8f;

// ---------------- problem constants ----------------
#define BATCH 64
#define SEQ   256
#define EMB   300
#define HID   300
#define HP    320      // padded hidden / embed (multiple of 32)
#define KT    10       // HP/32  k-tiles per 320
#define NT    20       // HP/16  n-tiles per 320
#define NTAGS 12

// ---------------- workspace layout (bytes) ----------------
// Wpack : bf16 B-fragments [dir][wt(ih,hh)][gate][jt=20][kt=10][512]
static constexpr size_t OFF_WPACK = 0;                 // 1,638,400 u16 = 3,276,800 B
static constexpr size_t OFF_BIAS  = 3276800;           // 2*4*320 f32   =    10,240 B
static constexpr size_t OFF_HA    = 3287040;           // [dir][buf2][mt=4][kt=10][512] u16 = 163,840 B
static constexpr size_t OFF_BAR   = 3450880;           // 64 u32        =       256 B
static constexpr size_t OFF_XA    = 3451136;           // [1024 mtiles][kt=10][512] u16 = 10,485,760 B
static constexpr size_t OFF_HOUT  = 13936896;          // [dir][s][b][304] f32 = 39,845,888 B
static constexpr size_t OFF_EMIT  = 53782784;          // [s][b][16] f32 = 1,048,576 B
// total ~54.8 MB

// ---------------- helpers ----------------
__device__ __forceinline__ unsigned short f32_to_bf16(float f) {
  unsigned u = __float_as_uint(f);
  u += 0x7FFFu + ((u >> 16) & 1u);        // round-to-nearest-even
  return (unsigned short)(u >> 16);
}

// A-fragment (16-bit, 16x32) element->K mapping: lane L, element e (of v16bf)
// K = (e/8)*16 + (L/16)*8 + ((e/2)%4)*2 + (e%2)   ; M = L%16
__device__ __forceinline__ int a_frag_k(int lane, int e) {
  return ((e >> 3) << 4) + ((lane >> 4) << 3) + (((e >> 1) & 3) << 1) + (e & 1);
}

__device__ __forceinline__ float sigmoidf(float x) { return 1.0f / (1.0f + __expf(-x)); }

__device__ __forceinline__ v16bf ldfrag(const unsigned short* p, int lane) {
  return *reinterpret_cast<const v16bf*>(p + lane * 16);
}

// 20-block release/acquire barrier (per direction) on L2 atomics.
// Arrival: scoped fetch-add; poll: scoped plain load (no RMW traffic).
__device__ __forceinline__ void grid_sync(unsigned* bar, unsigned nblk) {
  __threadfence();
  __syncthreads();
  if (threadIdx.x == 0) {
    unsigned* cnt = bar;
    unsigned* gen = bar + 16;
    unsigned g = __hip_atomic_load(gen, __ATOMIC_ACQUIRE, __HIP_MEMORY_SCOPE_AGENT);
    unsigned prev = __hip_atomic_fetch_add(cnt, 1u, __ATOMIC_ACQ_REL, __HIP_MEMORY_SCOPE_AGENT);
    if (prev == nblk - 1u) {
      __hip_atomic_store(cnt, 0u, __ATOMIC_RELAXED, __HIP_MEMORY_SCOPE_AGENT);
      __threadfence();
      __hip_atomic_fetch_add(gen, 1u, __ATOMIC_ACQ_REL, __HIP_MEMORY_SCOPE_AGENT);
    } else {
      while (__hip_atomic_load(gen, __ATOMIC_ACQUIRE, __HIP_MEMORY_SCOPE_AGENT) == g) {
        __builtin_amdgcn_s_sleep(1);
      }
    }
  }
  __syncthreads();
  __threadfence();
}

// ---------------- kernel 1: pack weights into B-fragment bf16 ----------------
// B tile (K x N = 32 x 16): element e of lane L -> K = (e/8)*16 + (L/16)*8 + (e%8), N = L%16
__global__ void prep_weights_k(const float* __restrict__ Wih_f, const float* __restrict__ Whh_f,
                               const float* __restrict__ Wih_b, const float* __restrict__ Whh_b,
                               unsigned short* __restrict__ wpack) {
  int idx = blockIdx.x * blockDim.x + threadIdx.x;          // 1,638,400 total
  int r    = idx & 511;
  int tile = idx >> 9;
  int lane = r >> 4, e = r & 15;
  int kt = tile % 10; int t1 = tile / 10;
  int jt = t1 % 20;   int t2 = t1 / 20;
  int g  = t2 % 4;    int t3 = t2 / 4;
  int wt = t3 % 2;    int dir = t3 / 2;
  int n = jt * 16 + (lane & 15);                            // output column (gate-local row of W)
  int K = kt * 32 + ((e >> 3) << 4) + ((lane >> 4) << 3) + (e & 7); // reduction index
  const float* W = (dir == 0) ? (wt ? Whh_f : Wih_f) : (wt ? Whh_b : Wih_b);
  float v = 0.0f;
  if (n < HID && K < HID) v = W[(g * HID + n) * EMB + K];   // W is (1200,300), EMB==HID==300
  wpack[idx] = f32_to_bf16(v);
}

// ---------------- kernel 2: zero h-state + pack bias + zero barrier ----------------
__global__ void prep_misc_k(const float* __restrict__ b_f, const float* __restrict__ b_b,
                            float* __restrict__ biasp, unsigned short* __restrict__ ha,
                            unsigned* __restrict__ bar) {
  int idx = blockIdx.x * blockDim.x + threadIdx.x;
  if (idx < 81920) { ha[idx] = 0; return; }
  int i2 = idx - 81920;
  if (i2 < 2560) {
    int dir = i2 / 1280; int r = i2 % 1280;
    int g = r / 320; int n = r % 320;
    const float* b = dir ? b_b : b_f;
    biasp[i2] = (n < HID) ? b[g * HID + n] : 0.0f;
    return;
  }
  int i3 = i2 - 2560;
  if (i3 < 64) bar[i3] = 0u;
}

// ---------------- kernel 3: embedding gather -> A-fragment bf16 ----------------
__global__ void embed_pack_k(const int* __restrict__ sent, const float* __restrict__ tab,
                             unsigned short* __restrict__ xa) {
  int idx = blockIdx.x * blockDim.x + threadIdx.x;          // 5,242,880 total
  int r    = idx & 511;
  int tile = idx >> 9;
  int lane = r >> 4, e = r & 15;
  int kt = tile % 10; int mt = tile / 10;                   // token M-tile 0..1023
  int token = mt * 16 + (lane & 15);
  int s = token >> 6, b = token & 63;
  int k = kt * 32 + a_frag_k(lane, e);
  float v = 0.0f;
  if (k < EMB) v = tab[(size_t)sent[b * SEQ + s] * EMB + k];
  xa[idx] = f32_to_bf16(v);
}

// ---------------- kernel 4: persistent bidirectional LSTM recurrence ----------------
// grid = 40 blocks x 128 threads. Block blk (0..19 per dir) owns hidden tile jt=blk;
// wave w owns batch tile mt=w. c-state lives in registers for all 256 steps; all 80
// weight B-fragments are loop-invariant and live in VGPRs (extended file via VGPR-MSB).
__global__ void __launch_bounds__(128) lstm_k(const unsigned short* __restrict__ wpack,
                                              const float* __restrict__ biasp,
                                              unsigned short* __restrict__ ha,
                                              const unsigned short* __restrict__ xa,
                                              float* __restrict__ hout,
                                              unsigned* __restrict__ bar) {
  const int dir  = blockIdx.x / 20;
  const int jt   = blockIdx.x % 20;
  const int mt   = threadIdx.x >> 5;          // wave id = batch tile
  const int lane = threadIdx.x & 31;
  const int hi   = lane >> 4;
  const int nloc = lane & 15;
  const int n    = jt * 16 + nloc;            // hidden column for C-layout lane
  unsigned* mybar = bar + dir * 32;

  // per-gate B-fragment bases for this jt
  const unsigned short* wih[4];
  const unsigned short* whh[4];
#pragma unroll
  for (int g = 0; g < 4; ++g) {
    wih[g] = wpack + (size_t)((((dir * 2 + 0) * 4 + g) * 20 + jt) * 10) * 512;
    whh[g] = wpack + (size_t)((((dir * 2 + 1) * 4 + g) * 20 + jt) * 10) * 512;
  }
  float bs[4];
#pragma unroll
  for (int g = 0; g < 4; ++g) bs[g] = biasp[dir * 1280 + g * 320 + n];

  // precompute h -> A-fragment scatter slot pieces (depend only on n)
  const int kt2  = n >> 5;
  const int kk   = n & 31;
  const int grp  = kk >> 4;
  const int rem  = kk & 15;
  const int hlfL = rem >> 3;
  const int r2   = rem & 7;
  const int eIdx = (grp * 4 + (r2 >> 1)) * 2 + (r2 & 1);

  v8f c = v8f{0.f, 0.f, 0.f, 0.f, 0.f, 0.f, 0.f, 0.f};

  for (int t = 0; t < SEQ; ++t) {
    const int s = dir ? (SEQ - 1 - t) : t;
    const int cur = t & 1;
    const unsigned short* xt = xa + (size_t)((s * 4 + mt) * 10) * 512;
    const unsigned short* ht = ha + (size_t)((((dir * 2 + cur) * 4) + mt) * 10) * 512;

    // ---- batch-issue ALL A-fragment loads for this step (one big clause) ----
    v16bf axf[KT];
    v16bf ahf[KT];
#pragma unroll
    for (int kt = 0; kt < KT; ++kt) axf[kt] = ldfrag(xt + kt * 512, lane);
#pragma unroll
    for (int kt = 0; kt < KT; ++kt) ahf[kt] = ldfrag(ht + kt * 512, lane);

    v8f acc[4];
#pragma unroll
    for (int g = 0; g < 4; ++g) acc[g] = v8f{0.f, 0.f, 0.f, 0.f, 0.f, 0.f, 0.f, 0.f};

    // x @ W_ih^T  (fused input transform)
#pragma unroll
    for (int kt = 0; kt < KT; ++kt) {
#pragma unroll
      for (int g = 0; g < 4; ++g) {
        v16bf b = ldfrag(wih[g] + kt * 512, lane);   // loop-invariant: hoisted to VGPRs
        acc[g] = __builtin_amdgcn_wmma_f32_16x16x32_bf16(false, axf[kt], false, b,
                                                         (short)0, acc[g], false, false);
      }
    }
    // h @ W_hh^T
#pragma unroll
    for (int kt = 0; kt < KT; ++kt) {
#pragma unroll
      for (int g = 0; g < 4; ++g) {
        v16bf b = ldfrag(whh[g] + kt * 512, lane);   // loop-invariant: hoisted to VGPRs
        acc[g] = __builtin_amdgcn_wmma_f32_16x16x32_bf16(false, ahf[kt], false, b,
                                                         (short)0, acc[g], false, false);
      }
    }

    // gates + state update; write h (f32 for emissions, bf16 A-frag for next step)
    const int nbuf = (t + 1) & 1;
    unsigned short* haN = ha + (size_t)((((dir * 2 + nbuf) * 4) + mt) * 10) * 512;
    float* hrow = hout + (size_t)((dir * SEQ + s) * BATCH) * 304;
#pragma unroll
    for (int v = 0; v < 8; ++v) {
      float gi = acc[0][v] + bs[0];
      float gf = acc[1][v] + bs[1];
      float gg = acc[2][v] + bs[2];
      float go = acc[3][v] + bs[3];
      float cn = sigmoidf(gf) * c[v] + sigmoidf(gi) * tanhf(gg);
      c[v] = cn;
      float h = sigmoidf(go) * tanhf(cn);
      int M = v + (hi << 3);                     // C layout: VGPR v, lanes>=16 -> M+8
      if (n < HID) hrow[(mt * 16 + M) * 304 + n] = h;
      haN[kt2 * 512 + (hlfL * 16 + M) * 16 + eIdx] = f32_to_bf16(h);
    }
    grid_sync(mybar, 20u);
  }
}

// ---------------- kernel 5: emission head (N=12, plain VALU) ----------------
__global__ void emit_k(const float* __restrict__ hout, const float* __restrict__ Wlin,
                       const float* __restrict__ blin, float* __restrict__ emit) {
  int token = blockIdx.x * blockDim.x + threadIdx.x;
  if (token >= SEQ * BATCH) return;
  int s = token >> 6, b = token & 63;
  const float* hf = hout + (size_t)((0 * SEQ + s) * BATCH + b) * 304;
  const float* hb = hout + (size_t)((1 * SEQ + s) * BATCH + b) * 304;
  float acc[NTAGS];
#pragma unroll
  for (int k = 0; k < NTAGS; ++k) acc[k] = blin[k];
  for (int j = 0; j < HID; ++j) {
    float xf = hf[j], xb = hb[j];
#pragma unroll
    for (int k = 0; k < NTAGS; ++k)
      acc[k] += Wlin[k * 600 + j] * xf + Wlin[k * 600 + 300 + j] * xb;
  }
  float* e = emit + (size_t)token * 16;
#pragma unroll
  for (int k = 0; k < NTAGS; ++k) e[k] = acc[k];
}

// ---------------- kernel 6: CRF forward + gold score ----------------
// one block, 768 threads = 64 batches x 12 tags
__global__ void crf_k(const float* __restrict__ emit, const int* __restrict__ tags,
                      const float* __restrict__ T, float* __restrict__ out) {
  __shared__ float Ts[NTAGS][NTAGS];
  __shared__ float ds[BATCH][NTAGS];
  int tid = threadIdx.x;
  if (tid < 144) Ts[tid / 12][tid % 12] = T[tid];
  int b = tid / 12, k = tid % 12;
  ds[b][k] = emit[(size_t)b * 16 + k];                 // s = 0
  __syncthreads();
  for (int s = 1; s < SEQ; ++s) {
    float m = -1e30f;
#pragma unroll
    for (int j = 0; j < NTAGS; ++j) m = fmaxf(m, ds[b][j] + Ts[j][k]);
    float sum = 0.0f;
#pragma unroll
    for (int j = 0; j < NTAGS; ++j) sum += __expf(ds[b][j] + Ts[j][k] - m);
    float dn = m + __logf(sum) + emit[(size_t)(s * BATCH + b) * 16 + k];
    __syncthreads();
    ds[b][k] = dn;
    __syncthreads();
  }
  if (k == 0) {
    float m = -1e30f;
#pragma unroll
    for (int j = 0; j < NTAGS; ++j) m = fmaxf(m, ds[b][j]);
    float sum = 0.0f;
#pragma unroll
    for (int j = 0; j < NTAGS; ++j) sum += __expf(ds[b][j] - m);
    float logz = m + __logf(sum);
    float tok = 0.0f, tr = 0.0f;
    int tprev = tags[b * SEQ];
    tok += emit[(size_t)(0 * BATCH + b) * 16 + tprev];
    for (int s = 1; s < SEQ; ++s) {
      int tc = tags[b * SEQ + s];
      tok += emit[(size_t)(s * BATCH + b) * 16 + tc];
      tr  += Ts[tprev][tc];
      tprev = tc;
    }
    out[b] = logz - tok - tr;
  }
}

// ---------------- launch ----------------
extern "C" void kernel_launch(void* const* d_in, const int* in_sizes, int n_in,
                              void* d_out, int out_size, void* d_ws, size_t ws_size,
                              hipStream_t stream) {
  (void)in_sizes; (void)n_in; (void)out_size; (void)ws_size;
  const int*   sentences = (const int*)  d_in[0];
  const int*   tags      = (const int*)  d_in[1];
  const float* embed     = (const float*)d_in[2];
  const float* Wih_f     = (const float*)d_in[3];
  const float* Whh_f     = (const float*)d_in[4];
  const float* b_f       = (const float*)d_in[5];
  const float* Wih_b     = (const float*)d_in[6];
  const float* Whh_b     = (const float*)d_in[7];
  const float* b_b       = (const float*)d_in[8];
  const float* Wlin      = (const float*)d_in[9];
  const float* blin      = (const float*)d_in[10];
  const float* trans     = (const float*)d_in[11];
  float* out = (float*)d_out;

  char* ws = (char*)d_ws;
  unsigned short* wpack = (unsigned short*)(ws + OFF_WPACK);
  float*          biasp = (float*)         (ws + OFF_BIAS);
  unsigned short* haBuf = (unsigned short*)(ws + OFF_HA);
  unsigned*       bar   = (unsigned*)      (ws + OFF_BAR);
  unsigned short* xaBuf = (unsigned short*)(ws + OFF_XA);
  float*          hout  = (float*)         (ws + OFF_HOUT);
  float*          emitB = (float*)         (ws + OFF_EMIT);

  prep_weights_k<<<6400, 256, 0, stream>>>(Wih_f, Whh_f, Wih_b, Whh_b, wpack);
  prep_misc_k   <<<331, 256, 0, stream>>>(b_f, b_b, biasp, haBuf, bar);
  embed_pack_k  <<<20480, 256, 0, stream>>>(sentences, embed, xaBuf);
  lstm_k        <<<40, 128, 0, stream>>>(wpack, biasp, haBuf, xaBuf, hout, bar);
  emit_k        <<<64, 256, 0, stream>>>(hout, Wlin, blin, emitB);
  crf_k         <<<1, 768, 0, stream>>>(emitB, tags, trans, out);
}
